// SelfAttentionLayer_90958817394855
// MI455X (gfx1250) — matrix operand order repaired and checked by reading, not compiled, not run on hardware
//
#include <hip/hip_runtime.h>
#include <hip/hip_bf16.h>
#include <math.h>
#include <stdint.h>

// Problem constants (match reference)
#define BB   4
#define SS   2048
#define DD   1024
#define HH   16
#define HDIM 64

typedef __attribute__((ext_vector_type(16))) __bf16 v16bf;
typedef __attribute__((ext_vector_type(8)))  __bf16 v8bf;
typedef __attribute__((ext_vector_type(8)))  float  v8f;
typedef __attribute__((ext_vector_type(4)))  unsigned int v4u;
typedef __attribute__((ext_vector_type(8)))  int v8i;
typedef __attribute__((ext_vector_type(4)))  int v4i;

__device__ __forceinline__ __bf16 f2bf(float f) {
  union { float f; unsigned u; } a; a.f = f;
  unsigned r = a.u + 0x7FFFu + ((a.u >> 16) & 1u);   // round-to-nearest-even
  union { unsigned short s; __bf16 b; } o; o.s = (unsigned short)(r >> 16);
  return o.b;
}

__device__ __forceinline__ v8f wmma_bf16(v16bf a, v16bf b, v8f c) {
  return __builtin_amdgcn_wmma_f32_16x16x32_bf16(false, a, false, b, (short)0, c,
                                                 false, false);
}

// ---------------------------------------------------------------- convert
__global__ void cvt_f32_bf16_kernel(const float* __restrict__ in,
                                    __bf16* __restrict__ out, size_t n) {
  size_t i = (size_t)blockIdx.x * blockDim.x + threadIdx.x;
  size_t stride = (size_t)gridDim.x * blockDim.x;
  for (; i < n; i += stride) out[i] = f2bf(in[i]);
}

// ---------------------------------------------------------------- GEMM
// C[M,N] = A[M,K](bf16,row-major) @ W[N,K](bf16,row-major)^T + bias
// 256 threads = 8 waves; wave tile 32x64 (2x4 WMMA tiles); WG tile 64x256.
// W tiles are staged into LDS by the Tensor Data Mover (double-buffered,
// 256 rows x 32 k per descriptor), overlapped with WMMA compute and tracked
// with TENSORcnt. A fragments stream from global (L0/L2-resident reuse).
enum { EPI_BF16 = 0, EPI_GELU = 1, EPI_RESF32 = 2 };

template <int EPI>
__global__ __launch_bounds__(256) void gemm_bf16_kernel(
    const __bf16* __restrict__ A, const __bf16* __restrict__ W,
    const float* __restrict__ bias, void* __restrict__ out,
    const float* __restrict__ res, int M, int N, int K) {
  __shared__ __bf16 wtile[2][256 * 32];  // 2 x 16 KB staging buffers
  const int lane = threadIdx.x & 31;
  const int wave = threadIdx.x >> 5;
  const int n16  = lane & 15;
  const int lhi  = lane >> 4;
  const int m_base = blockIdx.y * 64 + (wave & 1) * 32;
  const int n_base = blockIdx.x * 256;
  const int nw     = (wave >> 1) * 64;  // wave's n offset inside the 256 tile

  v8f acc[2][4] = {};

  // Issue one TDM descriptor: load W[n_base .. n_base+256) x [k0 .. k0+32)
  // (bf16) into wtile[buf]. D# built per CDNA5 ISA 8.3/8.4.
  auto stage = [&](int buf, int k0) {
#if __has_builtin(__builtin_amdgcn_tensor_load_to_lds)
    if (wave == 0) {
      unsigned long long ga =
          (unsigned long long)(uintptr_t)(W + (size_t)n_base * K + k0);
      unsigned lds = (unsigned)(uintptr_t)(&wtile[buf][0]);
      // group0: count=1 | lds_addr | global_addr[56:0] | type=2
      v4u g0 = { 1u, lds, (unsigned)ga,
                 (unsigned)((ga >> 32) & 0x1FFFFFFu) | (2u << 30) };
      // group1: data_size=2B; tensor_dim0=K, tensor_dim1=256 rows;
      //         tile_dim0=32, tile_dim1=256; tensor_dim0_stride=K
      v8i g1 = { (int)(1u << 16),
                 (int)(((unsigned)K & 0xFFFFu) << 16),
                 (int)(((unsigned)K >> 16) | (256u << 16)),
                 (int)(32u << 16),
                 (int)256,
                 (int)K,
                 0, 0 };
      v4i gz4 = { 0, 0, 0, 0 };
      v8i gz8 = { 0, 0, 0, 0, 0, 0, 0, 0 };
      __builtin_amdgcn_tensor_load_to_lds(g0, g1, gz4, gz4, gz8, 0);
    }
#else
    for (int i = threadIdx.x; i < 256 * 32 / 8; i += 256) {
      int row = (i * 8) / 32, col = (i * 8) % 32;
      *(v8bf*)&wtile[buf][row * 32 + col] =
          *(const v8bf*)(W + (size_t)(n_base + row) * K + k0 + col);
    }
#endif
  };

  stage(0, 0);
  const int KT = K / 32;
  for (int kt = 0; kt < KT; ++kt) {
    const int buf = kt & 1;
#if __has_builtin(__builtin_amdgcn_s_wait_tensorcnt)
    if (wave == 0) __builtin_amdgcn_s_wait_tensorcnt(0);
#endif
    __syncthreads();  // wtile[buf] ready; previous buffer's readers done
    if (kt + 1 < KT) stage(buf ^ 1, (kt + 1) * 32);

    const int k0 = kt * 32;
    v16bf afrag[2];
#pragma unroll
    for (int mt = 0; mt < 2; ++mt) {
      const __bf16* ap = A + (size_t)(m_base + mt * 16 + n16) * K + k0;
      v8bf lo = *(const v8bf*)(ap + lhi * 16);
      v8bf hi = *(const v8bf*)(ap + lhi * 16 + 8);
      afrag[mt] = __builtin_shufflevector(lo, hi, 0, 1, 2, 3, 4, 5, 6, 7, 8, 9,
                                          10, 11, 12, 13, 14, 15);
    }
    v16bf bfrag[4];
#pragma unroll
    for (int nt = 0; nt < 4; ++nt)
      bfrag[nt] =
          *(const v16bf*)&wtile[buf][(nw + nt * 16 + n16) * 32 + lhi * 16];
#pragma unroll
    for (int mt = 0; mt < 2; ++mt)
#pragma unroll
      for (int nt = 0; nt < 4; ++nt)
        acc[mt][nt] = wmma_bf16(afrag[mt], bfrag[nt], acc[mt][nt]);
  }

#pragma unroll
  for (int mt = 0; mt < 2; ++mt) {
#pragma unroll
    for (int nt = 0; nt < 4; ++nt) {
      const int n = n_base + nw + nt * 16 + n16;
      const float bv = bias[n];
#pragma unroll
      for (int r = 0; r < 8; ++r) {
        const int m = m_base + mt * 16 + lhi * 8 + r;
        const size_t idx = (size_t)m * N + n;
        float v = acc[mt][nt][r] + bv;
        if (EPI == EPI_BF16) {
          ((__bf16*)out)[idx] = f2bf(v);
        } else if (EPI == EPI_GELU) {
          float g = 0.5f * v * (1.0f + erff(v * 0.70710678118654752f));
          ((__bf16*)out)[idx] = f2bf(g);
        } else {
          ((float*)out)[idx] = v + res[idx];
        }
      }
    }
  }
}

// ---------------------------------------------------------------- V transpose
// vT[b][h][d][s] = qkv[b][s][2D + h*64 + d]   (bf16), 64x64 LDS tile
__global__ __launch_bounds__(256) void transpose_v_kernel(
    const __bf16* __restrict__ qkv, __bf16* __restrict__ vT) {
  __shared__ __bf16 tile[64 * 65];
  const int bid = blockIdx.x;
  const int st = bid & 31;  // S/64 = 32 tiles
  const int h  = (bid >> 5) & 15;
  const int b  = bid >> 9;
  const int s0 = st * 64;
  const int tid = threadIdx.x;
#pragma unroll
  for (int i = 0; i < 16; ++i) {
    int idx = tid + i * 256;
    int d = idx & 63, s = idx >> 6;
    tile[s * 65 + d] =
        qkv[(size_t)(b * SS + s0 + s) * (3 * DD) + 2 * DD + h * HDIM + d];
  }
  __syncthreads();
#pragma unroll
  for (int i = 0; i < 16; ++i) {
    int idx = tid + i * 256;
    int s = idx & 63, d = idx >> 6;
    vT[(((size_t)b * HH + h) * HDIM + d) * SS + s0 + s] = tile[s * 65 + d];
  }
}

// ---------------------------------------------------------------- attention
// One WG per (b, 32-row q tile); 128 threads = 4 waves.
// LDS: score tile 32x2048 fp32 (256 KB of the 320 KB/WGP) + reduction scratch.
// Loops all 16 heads so mean-attn rows are owned exclusively (no atomics).
__global__ __launch_bounds__(128) void attention_kernel(
    const __bf16* __restrict__ qkv, const __bf16* __restrict__ vT,
    __bf16* __restrict__ ctx, float* __restrict__ attnw) {
  extern __shared__ float smem[];
  float* sc  = smem;              // [32][2048]
  float* red = smem + 32 * 2048;  // [32][4]

  const int b   = blockIdx.x >> 6;  // S/32 = 64 q tiles
  const int q0  = (blockIdx.x & 63) * 32;
  const int tid = threadIdx.x;
  const int lane = tid & 31;
  const int wave = tid >> 5;
  const int n16  = lane & 15;
  const int lhi  = lane >> 4;
  const int srow  = tid & 31;  // softmax row ownership
  const int spart = tid >> 5;  // 4 parts of 512 cols

  for (int h = 0; h < HH; ++h) {
    const size_t qoff = (size_t)h * HDIM;

    // ---- scores = (Q @ K^T) / 8 -> LDS ----
    v16bf qfrag[2][2];  // [m tile][k step]
#pragma unroll
    for (int mt = 0; mt < 2; ++mt) {
#pragma unroll
      for (int kk = 0; kk < 2; ++kk) {
        const __bf16* qp = qkv +
            (size_t)(b * SS + q0 + mt * 16 + n16) * (3 * DD) + qoff + kk * 32;
        v8bf lo = *(const v8bf*)(qp + lhi * 16);
        v8bf hi = *(const v8bf*)(qp + lhi * 16 + 8);
        qfrag[mt][kk] = __builtin_shufflevector(lo, hi, 0, 1, 2, 3, 4, 5, 6, 7,
                                                8, 9, 10, 11, 12, 13, 14, 15);
      }
    }
    for (int nt = wave; nt < SS / 16; nt += 4) {
      const int n0 = nt * 16;
      v8f acc[2] = {};
#pragma unroll
      for (int kk = 0; kk < 2; ++kk) {
        const __bf16* kp = qkv + (size_t)(b * SS + n0 + n16) * (3 * DD) + DD +
                           qoff + kk * 32 + lhi * 16;
        v16bf bfrag = *(const v16bf*)kp;
        acc[0] = wmma_bf16(qfrag[0][kk], bfrag, acc[0]);
        acc[1] = wmma_bf16(qfrag[1][kk], bfrag, acc[1]);
      }
#pragma unroll
      for (int mt = 0; mt < 2; ++mt)
#pragma unroll
        for (int r = 0; r < 8; ++r)
          sc[(size_t)(mt * 16 + lhi * 8 + r) * SS + n0 + n16] =
              acc[mt][r] * 0.125f;
    }
    __syncthreads();

    // ---- softmax + mean-attn, vectorized float4 (4 threads/row, 512 cols) --
    {
      float4* row4 = (float4*)(sc + (size_t)srow * SS) + spart * 128;
      float mx = -3.4e38f;
      for (int c = 0; c < 128; ++c) {
        float4 v = row4[c];
        mx = fmaxf(mx, fmaxf(fmaxf(v.x, v.y), fmaxf(v.z, v.w)));
      }
      red[srow * 4 + spart] = mx;
      __syncthreads();
      float rmax = fmaxf(fmaxf(red[srow * 4 + 0], red[srow * 4 + 1]),
                         fmaxf(red[srow * 4 + 2], red[srow * 4 + 3]));
      __syncthreads();
      float s = 0.f;
      for (int c = 0; c < 128; ++c) {
        float4 v = row4[c];
        v.x = __expf(v.x - rmax);
        v.y = __expf(v.y - rmax);
        v.z = __expf(v.z - rmax);
        v.w = __expf(v.w - rmax);
        row4[c] = v;
        s += (v.x + v.y) + (v.z + v.w);
      }
      red[srow * 4 + spart] = s;
      __syncthreads();
      float inv = 1.0f / (red[srow * 4 + 0] + red[srow * 4 + 1] +
                          red[srow * 4 + 2] + red[srow * 4 + 3]);
      float4* aw4 =
          (float4*)(attnw + ((size_t)b * SS + q0 + srow) * SS) + spart * 128;
      const float mscale = 1.0f / 16.0f;
      for (int c = 0; c < 128; ++c) {
        float4 p = row4[c];
        p.x *= inv; p.y *= inv; p.z *= inv; p.w *= inv;
        row4[c] = p;
        float4 ct;
        ct.x = p.x * mscale; ct.y = p.y * mscale;
        ct.z = p.z * mscale; ct.w = p.w * mscale;
        if (h == 0) {
          aw4[c] = ct;  // exclusive rows: plain store, no atomics
        } else {
          float4 old = aw4[c];
          old.x += ct.x; old.y += ct.y; old.z += ct.z; old.w += ct.w;
          aw4[c] = old;
        }
      }
    }
    __syncthreads();

    // ---- ctx = P @ V (per wave: 1 m tile x 2 n tiles of the 32x64 tile) ----
    {
      const int mt  = wave & 1;
      const int ntb = (wave >> 1) * 2;
      v8f acc[2] = {};
      const float* prow = sc + (size_t)(mt * 16 + n16) * SS;
      for (int k0 = 0; k0 < SS; k0 += 32) {
        v16bf afrag;
#pragma unroll
        for (int i = 0; i < 8; ++i) afrag[i] = f2bf(prow[k0 + lhi * 16 + i]);
#pragma unroll
        for (int i = 0; i < 8; ++i)
          afrag[8 + i] = f2bf(prow[k0 + lhi * 16 + 8 + i]);
#pragma unroll
        for (int t = 0; t < 2; ++t) {
          const int d = (ntb + t) * 16 + n16;
          const __bf16* vp =
              vT + (((size_t)b * HH + h) * HDIM + d) * SS + k0 + lhi * 16;
          v16bf bfrag = *(const v16bf*)vp;
          acc[t] = wmma_bf16(afrag, bfrag, acc[t]);
        }
      }
#pragma unroll
      for (int t = 0; t < 2; ++t) {
        const int d = (ntb + t) * 16 + n16;
#pragma unroll
        for (int r = 0; r < 8; ++r) {
          const int q = q0 + mt * 16 + lhi * 8 + r;
          ctx[(size_t)(b * SS + q) * DD + h * HDIM + d] = f2bf(acc[t][r]);
        }
      }
    }
    __syncthreads();  // sc reused by next head
  }
}

// ---------------------------------------------------------------- layernorm
__global__ __launch_bounds__(128) void ln_kernel(
    const float* __restrict__ in, const float* __restrict__ g,
    const float* __restrict__ be, float* __restrict__ outF,
    __bf16* __restrict__ outB) {
  __shared__ float redb[128];
  const int row = blockIdx.x;
  const int tid = threadIdx.x;
  const float* x = in + (size_t)row * DD;
  float s = 0.f;
  for (int i = tid; i < DD; i += 128) s += x[i];
  redb[tid] = s;
  __syncthreads();
  for (int o = 64; o > 0; o >>= 1) {
    if (tid < o) redb[tid] += redb[tid + o];
    __syncthreads();
  }
  const float mean = redb[0] * (1.0f / DD);
  __syncthreads();
  float v = 0.f;
  for (int i = tid; i < DD; i += 128) {
    float d = x[i] - mean;
    v += d * d;
  }
  redb[tid] = v;
  __syncthreads();
  for (int o = 64; o > 0; o >>= 1) {
    if (tid < o) redb[tid] += redb[tid + o];
    __syncthreads();
  }
  const float rstd = rsqrtf(redb[0] * (1.0f / DD) + 1e-5f);
  for (int i = tid; i < DD; i += 128) {
    float y = (x[i] - mean) * rstd * g[i] + be[i];
    if (outF) outF[(size_t)row * DD + i] = y;
    if (outB) outB[(size_t)row * DD + i] = f2bf(y);
  }
}

// ---------------------------------------------------------------- launch
extern "C" void kernel_launch(void* const* d_in, const int* in_sizes, int n_in,
                              void* d_out, int out_size, void* d_ws,
                              size_t ws_size, hipStream_t stream) {
  (void)in_sizes; (void)n_in; (void)out_size; (void)ws_size;
  const float* features   = (const float*)d_in[0];
  const float* in_proj_w  = (const float*)d_in[1];
  const float* in_proj_b  = (const float*)d_in[2];
  const float* out_proj_w = (const float*)d_in[3];
  const float* out_proj_b = (const float*)d_in[4];
  const float* ln_g = (const float*)d_in[5];
  const float* ln_b = (const float*)d_in[6];
  const float* w1 = (const float*)d_in[7];
  const float* b1 = (const float*)d_in[8];
  const float* w2 = (const float*)d_in[9];
  const float* b2 = (const float*)d_in[10];

  float* out_f = (float*)d_out;                 // [B,S,D]
  float* attnw = out_f + (size_t)BB * SS * DD;  // [B,S,S]

  const size_t M = (size_t)BB * SS;  // 8192
  size_t off = 0;
  auto alloc = [&](size_t bytes) -> void* {
    off = (off + 255) & ~(size_t)255;
    void* p = (char*)d_ws + off;
    off += bytes;
    return p;
  };
  __bf16* xb    = (__bf16*)alloc(M * DD * 2);
  __bf16* wqkvb = (__bf16*)alloc((size_t)3 * DD * DD * 2);
  __bf16* woutb = (__bf16*)alloc((size_t)DD * DD * 2);
  __bf16* w1b   = (__bf16*)alloc((size_t)2 * DD * DD * 2);
  __bf16* w2b   = (__bf16*)alloc((size_t)DD * 2 * DD * 2);
  __bf16* qkvb  = (__bf16*)alloc(M * 3 * DD * 2);
  __bf16* vTb   = (__bf16*)alloc((size_t)BB * HH * HDIM * SS * 2);
  __bf16* ctxb  = (__bf16*)alloc(M * DD * 2);
  float*  s1    = (float*)alloc(M * DD * 4);
  float*  hf    = (float*)alloc(M * DD * 4);
  __bf16* hb    = (__bf16*)alloc(M * DD * 2);
  __bf16* g1b   = (__bf16*)alloc(M * 2 * DD * 2);
  float*  s2    = (float*)alloc(M * DD * 4);

  // fp32 -> bf16 conversions (activations + weights; weights stay hot in L2)
  cvt_f32_bf16_kernel<<<2048, 256, 0, stream>>>(features, xb, M * DD);
  cvt_f32_bf16_kernel<<<1024, 256, 0, stream>>>(in_proj_w, wqkvb,
                                                (size_t)3 * DD * DD);
  cvt_f32_bf16_kernel<<<512, 256, 0, stream>>>(out_proj_w, woutb,
                                               (size_t)DD * DD);
  cvt_f32_bf16_kernel<<<1024, 256, 0, stream>>>(w1, w1b, (size_t)2 * DD * DD);
  cvt_f32_bf16_kernel<<<1024, 256, 0, stream>>>(w2, w2b, (size_t)DD * 2 * DD);

  dim3 blk(256);
  // qkv = x @ Wqkv^T + b  -> bf16
  gemm_bf16_kernel<EPI_BF16><<<dim3(3 * DD / 256, M / 64), blk, 0, stream>>>(
      xb, wqkvb, in_proj_b, qkvb, nullptr, (int)M, 3 * DD, DD);

  transpose_v_kernel<<<BB * HH * (SS / 64), 256, 0, stream>>>(qkvb, vTb);

  const size_t smem = (32 * 2048 + 32 * 4) * sizeof(float);  // 256.5 KB
  (void)hipFuncSetAttribute((const void*)attention_kernel,
                            hipFuncAttributeMaxDynamicSharedMemorySize,
                            (int)smem);
  attention_kernel<<<BB * (SS / 32), 128, smem, stream>>>(qkvb, vTb, ctxb,
                                                          attnw);

  // attended + x  (residual fused)
  gemm_bf16_kernel<EPI_RESF32><<<dim3(DD / 256, M / 64), blk, 0, stream>>>(
      ctxb, woutb, out_proj_b, s1, features, (int)M, DD, DD);
  // h = LN(attended + x); keep fp32 (residual) + bf16 (GEMM input)
  ln_kernel<<<(int)M, 128, 0, stream>>>(s1, ln_g, ln_b, hf, hb);
  // gelu(h @ w1^T + b1) -> bf16
  gemm_bf16_kernel<EPI_GELU><<<dim3(2 * DD / 256, M / 64), blk, 0, stream>>>(
      hb, w1b, b1, g1b, nullptr, (int)M, 2 * DD, DD);
  // ff + h (residual fused)
  gemm_bf16_kernel<EPI_RESF32><<<dim3(DD / 256, M / 64), blk, 0, stream>>>(
      g1b, w2b, b2, s2, hf, (int)M, DD, 2 * DD);
  // out = LN(ff + h)
  ln_kernel<<<(int)M, 128, 0, stream>>>(s2, ln_g, ln_b, out_f, nullptr);
}
